// newffn_37623913513619
// MI455X (gfx1250) — matrix operand order repaired and checked by reading, not compiled, not run on hardware
//
#include <hip/hip_runtime.h>

// ---- problem constants (match reference) ----
#define T_TOK 4096      // B*S
#define DIM   1024      // D
#define NEXP  8         // experts
#define HEXP  4096      // expert hidden
#define HMLP  2048      // dense MLP hidden
#define GATE_BLOCKS 512 // 8 tokens per block

typedef __attribute__((ext_vector_type(16))) __bf16 v16bf;
typedef __attribute__((ext_vector_type(8)))  float  v8f;
typedef unsigned int u32;

// round-to-nearest-even fp32 -> bf16, packed pair into one dword
__device__ __forceinline__ u32 pack_bf16_2(float a, float b) {
    u32 ua = __float_as_uint(a), ub = __float_as_uint(b);
    u32 ra = (ua + 0x7fffu + ((ua >> 16) & 1u)) >> 16;
    u32 rb = (ub + 0x7fffu + ((ub >> 16) & 1u)) >> 16;
    return (ra & 0xffffu) | (rb << 16);
}

__device__ __forceinline__ unsigned short f32_to_bf16(float a) {
    u32 u = __float_as_uint(a);
    return (unsigned short)((u + 0x7fffu + ((u >> 16) & 1u)) >> 16);
}

// jax.nn.gelu default (approximate=True, tanh form)
__device__ __forceinline__ float gelu_tanh(float x) {
    const float k0 = 0.7978845608028654f;  // sqrt(2/pi)
    const float k1 = 0.044715f;
    float t = tanhf(k0 * (x + k1 * x * x * x));
    return 0.5f * x * (1.0f + t);
}

// ---------------------------------------------------------------------------
// One-time fp32 -> bf16 conversion of activations (no transpose)
// ---------------------------------------------------------------------------
__global__ void xconvert_kernel(const float* __restrict__ x, unsigned short* __restrict__ xbf) {
    int i = blockIdx.x * blockDim.x + threadIdx.x;       // pair index
    float2 v = ((const float2*)x)[i];
    ((u32*)xbf)[i] = pack_bf16_2(v.x, v.y);
}

// ---------------------------------------------------------------------------
// One-time weight convert + transpose: in [K,N] f32 (stacked on z) -> [N,K] bf16.
// 32x32 tiles through LDS so both global sides stay coalesced.
// ---------------------------------------------------------------------------
__launch_bounds__(256)
__global__ void wconvert_kernel(const float* __restrict__ W, unsigned short* __restrict__ Wt,
                                int K, int N) {
    const size_t zo = (size_t)blockIdx.z * K * N;
    const float* Wb = W + zo;
    unsigned short* Wo = Wt + zo;

    __shared__ float tile[32][33];
    const int n0 = blockIdx.x * 32, k0 = blockIdx.y * 32;
    const int c  = threadIdx.x & 31;       // n offset
    const int r0 = threadIdx.x >> 5;       // k offset base (stride 8)
#pragma unroll
    for (int i = 0; i < 4; ++i) {
        int r = r0 + i * 8;
        tile[r][c] = Wb[(size_t)(k0 + r) * N + n0 + c];
    }
    __syncthreads();
    const int kp = threadIdx.x & 15;       // dword col in out row
    const int n1 = threadIdx.x >> 4;       // 0..15
#pragma unroll
    for (int i = 0; i < 2; ++i) {
        int n = n1 + i * 16;
        u32* op = (u32*)(Wo + (size_t)(n0 + n) * K + k0) + kp;
        *op = pack_bf16_2(tile[2 * kp][n], tile[2 * kp + 1][n]);
    }
}

// ---------------------------------------------------------------------------
// Gating: one wave per token; deterministic per-block partials.
// ---------------------------------------------------------------------------
__launch_bounds__(256)
__global__ void gate_kernel(const float* __restrict__ x, const float* __restrict__ Wg,
                            int* __restrict__ topi, float* __restrict__ topw,
                            float* __restrict__ mePart, int* __restrict__ cntPart) {
    const int tid  = threadIdx.x;
    const int lane = tid & 31;
    const int wave = tid >> 5;
    const int t    = blockIdx.x * 8 + wave;

    float a[NEXP];
#pragma unroll
    for (int e = 0; e < NEXP; ++e) a[e] = 0.0f;

    const float* xr = x + (size_t)t * DIM;
    for (int d = lane; d < DIM; d += 32) {
        float xv = xr[d];
        const float* wr = Wg + d * NEXP;
#pragma unroll
        for (int e = 0; e < NEXP; ++e) a[e] += xv * wr[e];
    }
#pragma unroll
    for (int off = 16; off > 0; off >>= 1) {
#pragma unroll
        for (int e = 0; e < NEXP; ++e) a[e] += __shfl_xor(a[e], off, 32);
    }

    __shared__ float sMe[8][NEXP];
    __shared__ int   sCnt[8][NEXP];

    if (lane == 0) {
        float mx = a[0];
#pragma unroll
        for (int e = 1; e < NEXP; ++e) mx = fmaxf(mx, a[e]);
        float p[NEXP]; float s = 0.0f;
#pragma unroll
        for (int e = 0; e < NEXP; ++e) { p[e] = expf(a[e] - mx); s += p[e]; }
        float inv = 1.0f / s;
#pragma unroll
        for (int e = 0; e < NEXP; ++e) p[e] *= inv;
        int i0 = 0;
#pragma unroll
        for (int e = 1; e < NEXP; ++e) if (p[e] > p[i0]) i0 = e;
        int i1 = (i0 == 0) ? 1 : 0;
#pragma unroll
        for (int e = 0; e < NEXP; ++e) if (e != i0 && p[e] > p[i1]) i1 = e;
        float w0 = p[i0], w1 = p[i1];
        float wsum = w0 + w1;
        topi[t * 2 + 0] = i0;
        topi[t * 2 + 1] = i1;
        topw[t * 2 + 0] = w0 / wsum;
        topw[t * 2 + 1] = w1 / wsum;
#pragma unroll
        for (int e = 0; e < NEXP; ++e) {
            sMe[wave][e]  = p[e];
            sCnt[wave][e] = (e == i0 || e == i1) ? 1 : 0;
        }
    }
    __syncthreads();
    if (tid < NEXP) {
        float ms = 0.0f; int cs = 0;
#pragma unroll
        for (int w = 0; w < 8; ++w) { ms += sMe[w][tid]; cs += sCnt[w][tid]; }
        mePart[blockIdx.x * NEXP + tid]  = ms;
        cntPart[blockIdx.x * NEXP + tid] = cs;
    }
}

__global__ void zero_cnt_kernel(int* __restrict__ cnt) {
    if (threadIdx.x < NEXP) cnt[threadIdx.x] = 0;
}

__global__ void build_list_kernel(const int* __restrict__ topi, int* __restrict__ cnt,
                                  int* __restrict__ list) {
    int t = blockIdx.x * blockDim.x + threadIdx.x;
    if (t >= T_TOK) return;
#pragma unroll
    for (int s = 0; s < 2; ++s) {
        int e   = topi[t * 2 + s];
        int pos = atomicAdd(&cnt[e], 1);
        list[e * T_TOK + pos] = t * 2 + s;   // assignment id
    }
}

__global__ void loss_kernel(const float* __restrict__ lossIn,
                            const float* __restrict__ mePart,
                            const int* __restrict__ cntPart,
                            float* __restrict__ lossOut, int nblk) {
    __shared__ float sM[NEXP];
    __shared__ float sC[NEXP];
    int e = threadIdx.x;
    if (e < NEXP) {
        float ms = 0.0f; int cs = 0;
        for (int b = 0; b < nblk; ++b) { ms += mePart[b * NEXP + e]; cs += cntPart[b * NEXP + e]; }
        sM[e] = ms / (float)T_TOK;
        sC[e] = (float)cs / (float)(T_TOK * 2);
    }
    __syncthreads();
    if (threadIdx.x == 0) {
        float acc = 0.0f;
        for (int k = 0; k < NEXP; ++k) acc += sM[k] * sC[k];
        lossOut[0] = lossIn[0] + (float)NEXP * acc;
    }
}

// ---------------------------------------------------------------------------
// bf16 WMMA GEMM, all-bf16 operands:
//   C[64x64] = A[64xK](bf16, gathered rows) * Wt[NxK](bf16, pre-transposed) + bias
// 256 threads = 8 waves; wave w: n-subtile w&3, m-subtiles {w>>2, (w>>2)+2}.
// K chunked by 64 through double-buffered LDS; one barrier per chunk; next
// chunk prefetched into registers while current chunk computes.
// GATHER : rows via per-expert list (assignment id a = token*2+slot)
// ROWDIV2: A row = id>>1 (token) while output row = id (GEMM1 of MoE)
// GELU   : tanh-gelu + bf16 out, else f32 out
// ---------------------------------------------------------------------------
#define KP   36              // dword row stride in LDS (64 bf16 = 32 + 4 pad, 16B-aligned)
#define BUFS (64 * KP)       // dwords per buffer

template <int GATHER, int ROWDIV2, int GELU_OUT>
__launch_bounds__(256)
__global__ void ffn_gemm(const unsigned short* __restrict__ A,
                         const int* __restrict__ list,
                         const int* __restrict__ cnt,
                         const unsigned short* __restrict__ Wt,
                         const float* __restrict__ bias,
                         unsigned short* __restrict__ outBf,
                         float* __restrict__ outF,
                         int K, int N) {
    const int e  = GATHER ? blockIdx.z : 0;
    const int m0 = blockIdx.y * 64;
    const int n0 = blockIdx.x * 64;
    const int mCount = GATHER ? cnt[e] : T_TOK;
    if (m0 >= mCount) return;                       // block-uniform exit

    const int tid = threadIdx.x;

    // epilogue row-id table (valid-clamped, branch-free)
    __shared__ int sRow[64];
    if (tid < 64) {
        int m = m0 + tid;
        if (GATHER) {
            int mc = (m < mCount) ? m : (mCount - 1);
            sRow[tid] = list[e * T_TOK + mc];
        } else {
            sRow[tid] = m;
        }
    }

    __shared__ u32 sA[2 * BUFS];
    __shared__ u32 sB[2 * BUFS];

    // ---- staging assignment: each thread owns one row segment (8 dwords) ----
    const int sgRow = tid >> 2;                     // 0..63
    const int sgOff = (tid & 3) * 8;                // dword offset in row
    int rid;
    if (GATHER) {
        int m = m0 + sgRow;
        int mc = (m < mCount) ? m : (mCount - 1);
        rid = list[e * T_TOK + mc];
    } else {
        rid = m0 + sgRow;
    }
    const int arow = ROWDIV2 ? (rid >> 1) : rid;
    const u32* aPtr = (const u32*)(A + (size_t)arow * K) + sgOff;
    const u32* bPtr = (const u32*)(Wt + ((size_t)e * N + n0 + sgRow) * K) + sgOff;
    const int sgIdx = sgRow * KP + sgOff;

    // ---- per-wave fragment bases (both buffers) ----
    const int lane  = tid & 31;
    const int wave  = tid >> 5;
    const int nsub  = wave & 3;
    const int msub0 = wave >> 2;
    const int r16   = lane & 15;
    const int hi    = lane >> 4;

    const int aOff0 = (msub0 * 16 + r16) * KP + hi * 4;
    const int aOff1 = ((msub0 + 2) * 16 + r16) * KP + hi * 4;
    const int bOff  = (nsub * 16 + r16) * KP + hi * 4;

    v8f acc0 = {}; v8f acc1 = {};

    // preload chunk 0 into registers
    uint4 ra0 = *(const uint4*)(aPtr);
    uint4 ra1 = *(const uint4*)(aPtr + 4);
    uint4 rb0 = *(const uint4*)(bPtr);
    uint4 rb1 = *(const uint4*)(bPtr + 4);
    aPtr += 32; bPtr += 32;

    const int nch = K >> 6;
    for (int i = 0; i < nch; ++i) {
        const int p = i & 1;

        // store staged chunk into buffer p
        u32* as = sA + p * BUFS + sgIdx;
        u32* bs = sB + p * BUFS + sgIdx;
        *(uint4*)(as)     = ra0;
        *(uint4*)(as + 4) = ra1;
        *(uint4*)(bs)     = rb0;
        *(uint4*)(bs + 4) = rb1;

        // prefetch next chunk (overlaps barrier + WMMA below)
        if (i + 1 < nch) {
            ra0 = *(const uint4*)(aPtr);
            ra1 = *(const uint4*)(aPtr + 4);
            rb0 = *(const uint4*)(bPtr);
            rb1 = *(const uint4*)(bPtr + 4);
            aPtr += 32; bPtr += 32;
        }

        __syncthreads();   // buffer p stores visible; prior readers of p done

        const u32* aa0 = sA + p * BUFS + aOff0;
        const u32* aa1 = sA + p * BUFS + aOff1;
        const u32* ba  = sB + p * BUFS + bOff;
#pragma unroll
        for (int ks = 0; ks < 2; ++ks) {
            const int kb = ks * 16;
            union { u32 u[8]; v16bf v; } bf;
#pragma unroll
            for (int j = 0; j < 4; ++j) { bf.u[j] = ba[kb + j]; bf.u[4 + j] = ba[kb + 8 + j]; }

            union { u32 u[8]; v16bf v; } af;
#pragma unroll
            for (int j = 0; j < 4; ++j) { af.u[j] = aa0[kb + j]; af.u[4 + j] = aa0[kb + 8 + j]; }
            acc0 = __builtin_amdgcn_wmma_f32_16x16x32_bf16(false, af.v, false, bf.v,
                                                           (short)0, acc0, false, false);
#pragma unroll
            for (int j = 0; j < 4; ++j) { af.u[j] = aa1[kb + j]; af.u[4 + j] = aa1[kb + 8 + j]; }
            acc1 = __builtin_amdgcn_wmma_f32_16x16x32_bf16(false, af.v, false, bf.v,
                                                           (short)0, acc1, false, false);
        }
    }

    // ---- epilogue: lane(0-15): Vr -> M=r; lane(16-31): Vr -> M=8+r ----
    const int nCol = n0 + nsub * 16 + r16;
    const float bv = bias[(size_t)e * N + nCol];
    const int rowAdd = hi * 8;
#pragma unroll
    for (int half = 0; half < 2; ++half) {
        v8f acc = half ? acc1 : acc0;
        int mloc = (msub0 + 2 * half) * 16 + rowAdd;
#pragma unroll
        for (int r = 0; r < 8; ++r) {
            int m = m0 + mloc + r;
            if (GATHER && m >= mCount) continue;
            int orow = sRow[mloc + r];
            float v = acc[r] + bv;
            if (GELU_OUT) {
                outBf[(size_t)orow * N + nCol] = f32_to_bf16(gelu_tanh(v));
            } else {
                outF[(size_t)orow * N + nCol] = v;
            }
        }
    }
}

// out already holds y_mlp; add weighted expert outputs (fixed order)
__global__ void combine_kernel(const float* __restrict__ moeOut,
                               const float* __restrict__ topw,
                               float* __restrict__ y) {
    int i = blockIdx.x * blockDim.x + threadIdx.x;   // over T*D/4
    int t = i / (DIM / 4);
    int j = i % (DIM / 4);
    float w0 = topw[2 * t], w1 = topw[2 * t + 1];
    const float4* m0 = (const float4*)(moeOut + (size_t)(2 * t) * DIM) + j;
    const float4* m1 = (const float4*)(moeOut + (size_t)(2 * t + 1) * DIM) + j;
    float4* yp = (float4*)(y + (size_t)t * DIM) + j;
    float4 v = *yp, a = *m0, b = *m1;
    v.x += w0 * a.x + w1 * b.x;
    v.y += w0 * a.y + w1 * b.y;
    v.z += w0 * a.z + w1 * b.z;
    v.w += w0 * a.w + w1 * b.w;
    *yp = v;
}

extern "C" void kernel_launch(void* const* d_in, const int* in_sizes, int n_in,
                              void* d_out, int out_size, void* d_ws, size_t ws_size,
                              hipStream_t stream) {
    (void)in_sizes; (void)n_in; (void)out_size; (void)ws_size;

    const float* x    = (const float*)d_in[0];   // [T, D]
    const float* loss = (const float*)d_in[1];   // scalar
    const float* Wg   = (const float*)d_in[2];   // [D, E]
    const float* W1   = (const float*)d_in[3];   // [E, D, H]
    const float* b1   = (const float*)d_in[4];   // [E, H]
    const float* W2   = (const float*)d_in[5];   // [E, H, D]
    const float* b2   = (const float*)d_in[6];   // [E, D]
    const float* Wm1  = (const float*)d_in[7];   // [D, HM]
    const float* bm1  = (const float*)d_in[8];   // [HM]
    const float* Wm2  = (const float*)d_in[9];   // [HM, D]
    const float* bm2  = (const float*)d_in[10];  // [D]

    float* out     = (float*)d_out;              // [T*D] then loss scalar
    float* lossOut = out + (size_t)T_TOK * DIM;

    // workspace carve-out (~270 MB)
    char* ws = (char*)d_ws;
    size_t off = 0;
    auto alloc = [&](size_t bytes) -> void* {
        void* p = ws + off;
        off += (bytes + 255) & ~(size_t)255;
        return p;
    };
    int*   topi    = (int*)  alloc((size_t)T_TOK * 2 * sizeof(int));
    float* topw    = (float*)alloc((size_t)T_TOK * 2 * sizeof(float));
    float* mePart  = (float*)alloc((size_t)GATE_BLOCKS * NEXP * sizeof(float));
    int*   cntPart = (int*)  alloc((size_t)GATE_BLOCKS * NEXP * sizeof(int));
    int*   cnt     = (int*)  alloc((size_t)NEXP * sizeof(int));
    int*   list    = (int*)  alloc((size_t)NEXP * T_TOK * sizeof(int));
    unsigned short* xbf  = (unsigned short*)alloc((size_t)T_TOK * DIM * sizeof(unsigned short));
    unsigned short* W1t  = (unsigned short*)alloc((size_t)NEXP * DIM * HEXP * sizeof(unsigned short));
    unsigned short* W2t  = (unsigned short*)alloc((size_t)NEXP * HEXP * DIM * sizeof(unsigned short));
    unsigned short* Wm1t = (unsigned short*)alloc((size_t)DIM * HMLP * sizeof(unsigned short));
    unsigned short* Wm2t = (unsigned short*)alloc((size_t)HMLP * DIM * sizeof(unsigned short));
    unsigned short* hWs  = (unsigned short*)alloc((size_t)2 * T_TOK * HEXP * sizeof(unsigned short));
    unsigned short* hmWs = (unsigned short*)alloc((size_t)T_TOK * HMLP * sizeof(unsigned short));
    float* moeOut  = (float*)alloc((size_t)2 * T_TOK * DIM * sizeof(float));

    // 0) one-time precision/layout conversion (memory-bound)
    xconvert_kernel<<<(T_TOK * DIM / 2) / 256, 256, 0, stream>>>(x, xbf);
    { dim3 g(HEXP / 32, DIM / 32, NEXP);  wconvert_kernel<<<g, 256, 0, stream>>>(W1,  W1t,  DIM,  HEXP); }
    { dim3 g(DIM / 32, HEXP / 32, NEXP);  wconvert_kernel<<<g, 256, 0, stream>>>(W2,  W2t,  HEXP, DIM);  }
    { dim3 g(HMLP / 32, DIM / 32, 1);     wconvert_kernel<<<g, 256, 0, stream>>>(Wm1, Wm1t, DIM,  HMLP); }
    { dim3 g(DIM / 32, HMLP / 32, 1);     wconvert_kernel<<<g, 256, 0, stream>>>(Wm2, Wm2t, HMLP, DIM);  }

    // 1) routing
    zero_cnt_kernel<<<1, 32, 0, stream>>>(cnt);
    gate_kernel<<<GATE_BLOCKS, 256, 0, stream>>>(x, Wg, topi, topw, mePart, cntPart);
    build_list_kernel<<<T_TOK / 256, 256, 0, stream>>>(topi, cnt, list);
    loss_kernel<<<1, 32, 0, stream>>>(loss, mePart, cntPart, lossOut, GATE_BLOCKS);

    // 2) MoE expert FFN (routed)
    { dim3 g(HEXP / 64, T_TOK / 64, NEXP);
      ffn_gemm<1, 1, 1><<<g, 256, 0, stream>>>(xbf, list, cnt, W1t, b1, hWs, nullptr, DIM, HEXP); }
    { dim3 g(DIM / 64, T_TOK / 64, NEXP);
      ffn_gemm<1, 0, 0><<<g, 256, 0, stream>>>(hWs, list, cnt, W2t, b2, nullptr, moeOut, HEXP, DIM); }

    // 3) dense MLP
    { dim3 g(HMLP / 64, T_TOK / 64, 1);
      ffn_gemm<0, 0, 1><<<g, 256, 0, stream>>>(xbf, nullptr, nullptr, Wm1t, bm1, hmWs, nullptr, DIM, HMLP); }
    { dim3 g(DIM / 64, T_TOK / 64, 1);
      ffn_gemm<0, 0, 0><<<g, 256, 0, stream>>>(hmWs, nullptr, nullptr, Wm2t, bm2, nullptr, out, HMLP, DIM); }

    // 4) combine: out = y_mlp + w0*sel0 + w1*sel1
    combine_kernel<<<(T_TOK * (DIM / 4)) / 256, 256, 0, stream>>>(moeOut, topw, out);
}
